// GptOssSparseMoeBlock_84026740179802
// MI455X (gfx1250) — matrix operand order repaired and checked by reading, not compiled, not run on hardware
//
#include <hip/hip_runtime.h>

#define D_DIM   2048
#define F_DIM   2048
#define E_NUM   8
#define T_TOK   4096
#define NROWS   8192            // T_TOK * TOPK
#define ALPHA_C 1.702f
#define LIMIT_C 7.0f
#define AUX_COEF 0.01f
#define OUT_ELEMS 8388608       // 4*1024*2048

typedef __attribute__((ext_vector_type(16))) __bf16 v16bf;
typedef __attribute__((ext_vector_type(8)))  __bf16 v8bf;
typedef __attribute__((ext_vector_type(8)))  float  v8f;

// ---------------------------------------------------------------- zero
__global__ void zero_kernel(float4* __restrict__ out4, int n4,
                            int* __restrict__ counts, float* __restrict__ mean_probs) {
    int i = blockIdx.x * blockDim.x + threadIdx.x;
    float4 z = make_float4(0.f, 0.f, 0.f, 0.f);
    for (int k = i; k < n4; k += gridDim.x * blockDim.x) out4[k] = z;
    if (i < E_NUM) { counts[i] = 0; mean_probs[i] = 0.0f; }
}

// ---------------------------------------------------------------- router
// one wave (32 lanes) per token
__global__ void router_kernel(const float* __restrict__ x,
                              const float* __restrict__ rw,
                              const float* __restrict__ rb,
                              int* __restrict__ counts,
                              float* __restrict__ mean_probs,
                              int* __restrict__ tok_e,
                              float* __restrict__ tok_w) {
    int wave = threadIdx.x >> 5;
    int lane = threadIdx.x & 31;
    int t = blockIdx.x * 8 + wave;
    if (t >= T_TOK) return;
    const float* xr = x + (size_t)t * D_DIM;
    float acc[E_NUM];
#pragma unroll
    for (int e = 0; e < E_NUM; ++e) acc[e] = 0.0f;
    for (int d = lane; d < D_DIM; d += 32) {
        float xv = xr[d];
        const float* w = rw + (size_t)d * E_NUM;
#pragma unroll
        for (int e = 0; e < E_NUM; ++e) acc[e] += xv * w[e];
    }
#pragma unroll
    for (int e = 0; e < E_NUM; ++e)
        for (int off = 16; off > 0; off >>= 1)
            acc[e] += __shfl_xor(acc[e], off, 32);
    if (lane == 0) {
        float lg[E_NUM], pr[E_NUM];
        float mx = -1e30f;
#pragma unroll
        for (int e = 0; e < E_NUM; ++e) { lg[e] = acc[e] + rb[e]; mx = fmaxf(mx, lg[e]); }
        float se = 0.0f;
#pragma unroll
        for (int e = 0; e < E_NUM; ++e) { pr[e] = __expf(lg[e] - mx); se += pr[e]; }
        float inv = 1.0f / se;
#pragma unroll
        for (int e = 0; e < E_NUM; ++e)
            atomicAdd(&mean_probs[e], pr[e] * inv * (1.0f / (float)T_TOK));
        int e0 = 0;
#pragma unroll
        for (int e = 1; e < E_NUM; ++e) if (lg[e] > lg[e0]) e0 = e;
        int e1 = (e0 == 0) ? 1 : 0;
#pragma unroll
        for (int e = 0; e < E_NUM; ++e) { if (e == e0 || e == e1) continue; if (lg[e] > lg[e1]) e1 = e; }
        float w0 = 1.0f / (1.0f + __expf(lg[e1] - lg[e0]));
        float w1 = 1.0f - w0;
        tok_e[t * 2 + 0] = e0; tok_e[t * 2 + 1] = e1;
        tok_w[t * 2 + 0] = w0; tok_w[t * 2 + 1] = w1;
        atomicAdd(&counts[e0], 1);
        atomicAdd(&counts[e1], 1);
    }
}

// ---------------------------------------------------------------- offsets + aux stats
__global__ void stats_kernel(const int* __restrict__ counts,
                             const float* __restrict__ mean_probs,
                             int* __restrict__ offsets,
                             float* __restrict__ out_tail) {
    if (threadIdx.x == 0) {
        int off = 0; float loss = 0.0f;
        for (int e = 0; e < E_NUM; ++e) {
            offsets[e] = off; off += counts[e];
            float df = (float)counts[e] / (float)(T_TOK * 2);
            out_tail[e] = df;
            loss += df * mean_probs[e];
        }
        out_tail[E_NUM] = AUX_COEF * (float)E_NUM * loss;
    }
}

// ---------------------------------------------------------------- deterministic row assignment
__global__ void build_rows_kernel(const int* __restrict__ offsets,
                                  const int* __restrict__ tok_e,
                                  const float* __restrict__ tok_w,
                                  int* __restrict__ row_tok,
                                  float* __restrict__ row_w) {
    int e = blockIdx.x;
    __shared__ int sflags[256];
    int base = offsets[e];
    for (int c0 = 0; c0 < T_TOK; c0 += 256) {
        int t = c0 + threadIdx.x;
        int slot = -1;
        if (tok_e[t * 2 + 0] == e) slot = 0;
        else if (tok_e[t * 2 + 1] == e) slot = 1;
        int f = (slot >= 0) ? 1 : 0;
        sflags[threadIdx.x] = f;
        __syncthreads();
        for (int off = 1; off < 256; off <<= 1) {
            int tmp = (threadIdx.x >= off) ? sflags[threadIdx.x - off] : 0;
            __syncthreads();
            sflags[threadIdx.x] += tmp;
            __syncthreads();
        }
        int incl = sflags[threadIdx.x];
        int total = sflags[255];
        if (slot >= 0) {
            int r = base + incl - 1;
            row_tok[r] = t;
            row_w[r] = tok_w[t * 2 + slot];
        }
        base += total;
        __syncthreads();
    }
}

// ---------------------------------------------------------------- gather x -> bf16 rows
__global__ void gather_kernel(const float* __restrict__ x,
                              const int* __restrict__ row_tok,
                              __bf16* __restrict__ Xg) {
    int r = blockIdx.x;
    int t = row_tok[r];
    const float4* src = (const float4*)(x + (size_t)t * D_DIM);
    v8bf* dst = (v8bf*)(Xg + (size_t)r * D_DIM);
    for (int i = threadIdx.x; i < D_DIM / 8; i += blockDim.x) {
        float4 a = src[i * 2 + 0];
        float4 b = src[i * 2 + 1];
        v8bf o;
        o[0] = (__bf16)a.x; o[1] = (__bf16)a.y; o[2] = (__bf16)a.z; o[3] = (__bf16)a.w;
        o[4] = (__bf16)b.x; o[5] = (__bf16)b.y; o[6] = (__bf16)b.z; o[7] = (__bf16)b.w;
        dst[i] = o;
    }
}

// ---------------------------------------------------------------- WMMA mainloop
// Block tile: 128(M) x 128(N), K staged 64-wide. 8 waves: 4(M) x 2(N),
// each wave 32x64 = 2x4 tiles of v_wmma_f32_16x16x32_bf16.
// A (bf16): moved global->LDS with CDNA5 async-DMA (global_load_async_to_lds_b128,
// ASYNCcnt). B (fp32 weights, row-major KxN): converted to bf16 while staging
// transposed into LDS (Blds[n][k]).
template <int NSTRIDE>
__device__ __forceinline__ void gemm_mainloop(const __bf16* __restrict__ Asrc,
                                              const float* __restrict__ B,
                                              int row0, int n0,
                                              __bf16 (*Alds)[72], __bf16 (*Blds)[72],
                                              v8f acc[2][4]) {
    const int tid  = threadIdx.x;
    const int lane = tid & 31;
    const int wave = tid >> 5;
    const int wm = wave >> 1, wn = wave & 1;
    const int l15 = lane & 15, hi = lane >> 4;

    // per-thread A-tile slice: 32 bf16 = 4 x b128 async DMAs
    const int am  = tid >> 1;
    const int akh = (tid & 1) * 32;
    long agr = row0 + am; if (agr > NROWS - 1) agr = NROWS - 1;
    const __bf16* agsrc0 = Asrc + (size_t)agr * 2048 + akh;
    const unsigned albase = (unsigned)(unsigned long long)&Alds[am][akh];

    for (int k0 = 0; k0 < 2048; k0 += 64) {
        __syncthreads();
        // ---- A tile: 128 rows x 64 k, async DMA straight into LDS
        {
            unsigned long long ga = (unsigned long long)(agsrc0 + k0);
#pragma unroll
            for (int j = 0; j < 4; ++j) {
                asm volatile("global_load_async_to_lds_b128 %0, %1, off"
                             :: "v"(albase + j * 16u), "v"(ga + (unsigned long long)(j * 16))
                             : "memory");
            }
        }
        // ---- B tile transposed: fp32 -> bf16, Blds[n][k]
        {
#pragma unroll
            for (int j = 0; j < 8; ++j) {
                int c  = tid + 256 * j;
                int kk = c >> 5;
                int nc = (c & 31) * 4;
                const float* gp = B + (size_t)(k0 + kk) * NSTRIDE + n0 + nc;
                if (j == 0 && k0 + 64 < 2048)
                    __builtin_prefetch(gp + (size_t)64 * NSTRIDE, 0, 1);
                float4 w = *(const float4*)gp;
                Blds[nc + 0][kk] = (__bf16)w.x;
                Blds[nc + 1][kk] = (__bf16)w.y;
                Blds[nc + 2][kk] = (__bf16)w.z;
                Blds[nc + 3][kk] = (__bf16)w.w;
            }
        }
        // wait for this wave's async DMAs, then make all waves' stages visible
        asm volatile("s_wait_asynccnt 0x0" ::: "memory");
        __syncthreads();
        // ---- compute: two k-substeps of 32
#pragma unroll
        for (int ks = 0; ks < 64; ks += 32) {
            v16bf af[2], bfr[4];
#pragma unroll
            for (int mi = 0; mi < 2; ++mi) {
                int ml = wm * 32 + mi * 16 + l15;
                v8bf a0 = *(const v8bf*)&Alds[ml][ks + 8 * hi];
                v8bf a1 = *(const v8bf*)&Alds[ml][ks + 16 + 8 * hi];
#pragma unroll
                for (int q = 0; q < 8; ++q) { af[mi][q] = a0[q]; af[mi][q + 8] = a1[q]; }
            }
#pragma unroll
            for (int ni = 0; ni < 4; ++ni) {
                int nl = wn * 64 + ni * 16 + l15;
                v8bf b0 = *(const v8bf*)&Blds[nl][ks + 8 * hi];
                v8bf b1 = *(const v8bf*)&Blds[nl][ks + 16 + 8 * hi];
#pragma unroll
                for (int q = 0; q < 8; ++q) { bfr[ni][q] = b0[q]; bfr[ni][q + 8] = b1[q]; }
            }
#pragma unroll
            for (int mi = 0; mi < 2; ++mi)
#pragma unroll
                for (int ni = 0; ni < 4; ++ni)
                    acc[mi][ni] = __builtin_amdgcn_wmma_f32_16x16x32_bf16(
                        false, af[mi], false, bfr[ni], (short)0, acc[mi][ni], false, false);
        }
    }
}

// ---------------------------------------------------------------- GEMM1: gate_up + activation
__global__ void __launch_bounds__(256)
gemm1_kernel(const __bf16* __restrict__ Xg,
             const float* __restrict__ w_gate_up,
             const float* __restrict__ b_gate_up,
             const int* __restrict__ counts,
             const int* __restrict__ offsets,
             __bf16* __restrict__ Act) {
    __shared__ __align__(16) char smem[2 * 128 * 72 * 2];
    __bf16 (*Alds)[72] = (__bf16(*)[72])smem;
    __bf16 (*Blds)[72] = (__bf16(*)[72])(smem + 128 * 72 * 2);

    int e  = blockIdx.x >> 5;
    int mt = blockIdx.x & 31;
    int cnt = counts[e], off = offsets[e];
    if (mt * 128 >= cnt) return;
    int nt = blockIdx.y;                 // 32 tiles over 2F=4096
    int row0 = off + mt * 128;
    const float* B = w_gate_up + (size_t)e * D_DIM * (2 * F_DIM);

    v8f zed = {0.f, 0.f, 0.f, 0.f, 0.f, 0.f, 0.f, 0.f};
    v8f acc[2][4];
#pragma unroll
    for (int mi = 0; mi < 2; ++mi)
#pragma unroll
        for (int ni = 0; ni < 4; ++ni) acc[mi][ni] = zed;

    gemm_mainloop<2 * F_DIM>(Xg, B, row0, nt * 128, Alds, Blds, acc);

    // ---- fused epilogue: LDS round-trip, pair (gate, up), SiLU-gate, store bf16
    float (*Clds)[68] = (float(*)[68])smem;
    const int lane = threadIdx.x & 31, wave = threadIdx.x >> 5;
    const int wm = wave >> 1, wn = wave & 1, l15 = lane & 15, hi = lane >> 4;
    const float* bias = b_gate_up + (size_t)e * (2 * F_DIM);

    for (int half = 0; half < 2; ++half) {
        __syncthreads();
        if (wn == half) {
#pragma unroll
            for (int mi = 0; mi < 2; ++mi)
#pragma unroll
                for (int ni = 0; ni < 4; ++ni) {
                    int nl = ni * 16 + l15;
#pragma unroll
                    for (int q = 0; q < 8; ++q) {
                        int ml = wm * 32 + mi * 16 + q + 8 * hi;
                        Clds[ml][nl] = acc[mi][ni][q];
                    }
                }
        }
        __syncthreads();
        int f0 = nt * 64 + half * 32;          // global act-column base
        for (int idx = threadIdx.x; idx < 128 * 4; idx += 256) {
            int m  = idx >> 2;
            int fg = (idx & 3) * 8;
            int rl = mt * 128 + m;
            if (rl >= cnt) continue;
            v8bf o;
#pragma unroll
            for (int q = 0; q < 8; ++q) {
                int fi = fg + q;
                int gub = nt * 128 + half * 64 + fi * 2;
                float gate = Clds[m][fi * 2 + 0] + bias[gub + 0];
                float up   = Clds[m][fi * 2 + 1] + bias[gub + 1];
                gate = fminf(gate, LIMIT_C);
                up   = fminf(fmaxf(up, -LIMIT_C), LIMIT_C);
                float glu = gate / (1.0f + __expf(-gate * ALPHA_C));
                o[q] = (__bf16)((up + 1.0f) * glu);
            }
            *(v8bf*)&Act[(size_t)(row0 + m) * F_DIM + f0 + fg] = o;
        }
    }
}

// ---------------------------------------------------------------- GEMM2: down proj + scatter
__global__ void __launch_bounds__(256)
gemm2_kernel(const __bf16* __restrict__ Act,
             const float* __restrict__ w_down,
             const float* __restrict__ b_down,
             const int* __restrict__ counts,
             const int* __restrict__ offsets,
             const int* __restrict__ row_tok,
             const float* __restrict__ row_w,
             float* __restrict__ out) {
    __shared__ __align__(16) char smem[2 * 128 * 72 * 2];
    __bf16 (*Alds)[72] = (__bf16(*)[72])smem;
    __bf16 (*Blds)[72] = (__bf16(*)[72])(smem + 128 * 72 * 2);

    int e  = blockIdx.x >> 5;
    int mt = blockIdx.x & 31;
    int cnt = counts[e], off = offsets[e];
    if (mt * 128 >= cnt) return;
    int n0 = blockIdx.y * 128;            // 16 tiles over D=2048
    int row0 = off + mt * 128;
    const float* B = w_down + (size_t)e * F_DIM * D_DIM;

    v8f zed = {0.f, 0.f, 0.f, 0.f, 0.f, 0.f, 0.f, 0.f};
    v8f acc[2][4];
#pragma unroll
    for (int mi = 0; mi < 2; ++mi)
#pragma unroll
        for (int ni = 0; ni < 4; ++ni) acc[mi][ni] = zed;

    gemm_mainloop<D_DIM>(Act, B, row0, n0, Alds, Blds, acc);

    const int lane = threadIdx.x & 31, wave = threadIdx.x >> 5;
    const int wm = wave >> 1, wn = wave & 1, l15 = lane & 15, hi = lane >> 4;
    const float* bias = b_down + (size_t)e * D_DIM;

#pragma unroll
    for (int mi = 0; mi < 2; ++mi)
#pragma unroll
        for (int q = 0; q < 8; ++q) {
            int m  = wm * 32 + mi * 16 + q + 8 * hi;
            int rl = mt * 128 + m;
            if (rl >= cnt) continue;
            int r = row0 + m;
            int t = row_tok[r];
            float w = row_w[r];
            float* orow = out + (size_t)t * D_DIM;
#pragma unroll
            for (int ni = 0; ni < 4; ++ni) {
                int n = n0 + wn * 64 + ni * 16 + l15;
                atomicAdd(orow + n, (acc[mi][ni][q] + bias[n]) * w);
            }
        }
}

// ---------------------------------------------------------------- host launcher
extern "C" void kernel_launch(void* const* d_in, const int* in_sizes, int n_in,
                              void* d_out, int out_size, void* d_ws, size_t ws_size,
                              hipStream_t stream) {
    const float* x         = (const float*)d_in[0];
    const float* router_w  = (const float*)d_in[1];
    const float* router_b  = (const float*)d_in[2];
    const float* w_gate_up = (const float*)d_in[3];
    const float* b_gate_up = (const float*)d_in[4];
    const float* w_down    = (const float*)d_in[5];
    const float* b_down    = (const float*)d_in[6];
    float* out = (float*)d_out;
    (void)in_sizes; (void)n_in; (void)out_size; (void)ws_size;

    // workspace carve-out (all sections 256B-aligned)
    char* p = (char*)d_ws;
    int*    counts     = (int*)p;    p += 256;
    int*    offsets    = (int*)p;    p += 256;
    float*  mean_probs = (float*)p;  p += 256;
    int*    tok_e      = (int*)p;    p += T_TOK * 2 * sizeof(int);
    float*  tok_w      = (float*)p;  p += T_TOK * 2 * sizeof(float);
    int*    row_tok    = (int*)p;    p += NROWS * sizeof(int);
    float*  row_w      = (float*)p;  p += NROWS * sizeof(float);
    __bf16* Xg         = (__bf16*)p; p += (size_t)NROWS * D_DIM * sizeof(__bf16);
    __bf16* Act        = (__bf16*)p;

    // note: the 9 stats tail elements of d_out are fully overwritten by stats_kernel
    zero_kernel<<<4096, 256, 0, stream>>>((float4*)out, OUT_ELEMS / 4, counts, mean_probs);
    router_kernel<<<T_TOK / 8, 256, 0, stream>>>(x, router_w, router_b,
                                                 counts, mean_probs, tok_e, tok_w);
    stats_kernel<<<1, 32, 0, stream>>>(counts, mean_probs, offsets, out + OUT_ELEMS);
    build_rows_kernel<<<E_NUM, 256, 0, stream>>>(offsets, tok_e, tok_w, row_tok, row_w);
    gather_kernel<<<NROWS, 256, 0, stream>>>(x, row_tok, Xg);
    gemm1_kernel<<<dim3(E_NUM * 32, 32), 256, 0, stream>>>(Xg, w_gate_up, b_gate_up,
                                                           counts, offsets, Act);
    gemm2_kernel<<<dim3(E_NUM * 32, 16), 256, 0, stream>>>(Act, w_down, b_down,
                                                           counts, offsets,
                                                           row_tok, row_w, out);
}
